// Kan_Block_54941221650461
// MI455X (gfx1250) — matrix operand order repaired and checked by reading, not compiled, not run on hardware
//
#include <hip/hip_runtime.h>
#include <hip/hip_bf16.h>

// ---------------------------------------------------------------------------
// KAN block (2x KAN-linear + GELU + residual) for MI455X / gfx1250.
// Base+spline paths folded into one bf16 WMMA GEMM per layer via augmented
// activations [silu(x), B0..B7(x)] and augmented K-major bf16 weights.
// GEMM waves compute 32x64 tiles (2 A-frags x 4 B-frags -> 8 WMMA / K-step)
// with ping-pong double-buffered fragment loads (2x-unrolled K loop, no
// register rotation copies, partial s_wait_loadcnt only).
// ---------------------------------------------------------------------------

typedef __attribute__((ext_vector_type(16))) __bf16 v16bf;
typedef __attribute__((ext_vector_type(8)))  __bf16 v8bf;
typedef __attribute__((ext_vector_type(8)))  float  v8f;

#define KAN_GRID  5
#define KAN_ORDER 3
#define NCOEF     8   // KAN_GRID + KAN_ORDER
#define NAUG      9   // 1 (silu/base) + NCOEF (spline basis)

__device__ __forceinline__ float siluf(float x) {
  return x / (1.0f + __expf(-x));
}
__device__ __forceinline__ float geluf(float x) {
  return 0.5f * x * (1.0f + erff(x * 0.70710678118654752f));
}

// Cubic B-spline basis on the fixed uniform grid of the reference:
// knots t[j] = -1 + (j-3)*0.4, j=0..11; returns 8 basis values in b[0..7].
__device__ __forceinline__ void bspline8(float x, float b[11]) {
  const float h = 2.0f / KAN_GRID; // 0.4
  float t[12];
#pragma unroll
  for (int j = 0; j < 12; ++j) t[j] = -1.0f + (float)(j - KAN_ORDER) * h;
#pragma unroll
  for (int j = 0; j < 11; ++j) b[j] = (x >= t[j] && x < t[j + 1]) ? 1.0f : 0.0f;
#pragma unroll
  for (int k = 1; k <= KAN_ORDER; ++k) {
    float inv = 1.0f / ((float)k * h);
#pragma unroll
    for (int j = 0; j < 11 - k; ++j)
      b[j] = (x - t[j]) * inv * b[j] + (t[j + k + 1] - x) * inv * b[j + 1];
  }
}

// ---------------------------------------------------------------------------
// Pack augmented weights, K-major: Wt[(i*9+j)*OUT + o], bf16.
// ---------------------------------------------------------------------------
__global__ __launch_bounds__(256)
void pack_wt(const float* __restrict__ base_w,
             const float* __restrict__ spline_w,
             const float* __restrict__ scaler,
             __bf16* __restrict__ Wt, int IN, int OUT) {
  int o = blockIdx.x * blockDim.x + threadIdx.x;
  int i = blockIdx.y;
  if (o >= OUT) return;
  size_t wi = (size_t)o * IN + i;
  float bw = base_w[wi];
  float sc = scaler[wi];
  const float* sp = spline_w + wi * NCOEF;
  size_t kbase = (size_t)i * NAUG;
  Wt[kbase * OUT + o] = (__bf16)bw;
#pragma unroll
  for (int j = 0; j < NCOEF; ++j)
    Wt[(kbase + 1 + j) * OUT + o] = (__bf16)(sp[j] * sc);
}

// ---------------------------------------------------------------------------
// Augmented activations, row-major: A[n, i*9+j] = [silu(x), basis0..7], bf16.
// ---------------------------------------------------------------------------
__global__ __launch_bounds__(256)
void augment_act(const float* __restrict__ X, __bf16* __restrict__ A, int IN) {
  int i = blockIdx.x * blockDim.x + threadIdx.x;
  int n = blockIdx.y;
  if (i >= IN) return;
  float v = X[(size_t)n * IN + i];
  __bf16* dst = A + ((size_t)n * IN + i) * NAUG;
  dst[0] = (__bf16)siluf(v);
  float b[11];
  bspline8(v, b);
#pragma unroll
  for (int j = 0; j < NCOEF; ++j) dst[1 + j] = (__bf16)b[j];
}

// ---------------------------------------------------------------------------
// WMMA main loop, ping-pong double buffered.
// A fragment (ISA 16-bit A 16x32 layout): lane L<16 -> row m+L holds
// K {kb..kb+7, kb+16..kb+23}; lane L>=16 -> row m+(L-16) holds
// K {kb+8..kb+15, kb+24..kb+31}.
// B fragment (K-major Wt): lane L holds column N=n+(L&15), K in the same order as A.
// ---------------------------------------------------------------------------
__device__ __forceinline__ v16bf cat8(v8bf a, v8bf b) {
  return __builtin_shufflevector(a, b, 0, 1, 2, 3, 4, 5, 6, 7,
                                 8, 9, 10, 11, 12, 13, 14, 15);
}
__device__ __forceinline__ v16bf ldfrag(const __bf16* p0, const __bf16* p1) {
  return cat8(*(const v8bf*)p0, *(const v8bf*)p1);
}

struct Frags {
  v16bf a0, a1, b[4];
};

__device__ __forceinline__ void ldfrags(Frags& f, const __bf16* arow0,
                                        const __bf16* arow1,
                                        const __bf16* bbase, int kb, int OUT) {
  f.a0 = ldfrag(arow0 + kb, arow0 + kb + 16);
  f.a1 = ldfrag(arow1 + kb, arow1 + kb + 16);
  const __bf16* brow = bbase + (size_t)kb * OUT;
#pragma unroll
  for (int t = 0; t < 4; ++t)
#pragma unroll
    for (int s = 0; s < 16; ++s)
      f.b[t][s] = brow[(size_t)(s < 8 ? s : s + 8) * OUT + t * 16];
}

__device__ __forceinline__ void do_wmma(const Frags& f, v8f acc[2][4]) {
#pragma unroll
  for (int t = 0; t < 4; ++t) {
    acc[0][t] = __builtin_amdgcn_wmma_f32_16x16x32_bf16(
        false, f.a0, false, f.b[t], (short)0, acc[0][t], false, false);
    acc[1][t] = __builtin_amdgcn_wmma_f32_16x16x32_bf16(
        false, f.a1, false, f.b[t], (short)0, acc[1][t], false, false);
  }
}

// K must be a multiple of 64 (K1 = 9216, K2 = 36864: both are).
__device__ __forceinline__ void wmma_mainloop(const __bf16* __restrict__ A,
                                              const __bf16* __restrict__ Wt,
                                              int K, int OUT, int m0, int n0,
                                              int lane, v8f acc[2][4]) {
  int halfsel = (lane >> 4) * 8;                       // 0 or 8
  const __bf16* arow0 = A + (size_t)(m0 + (lane & 15)) * K + halfsel;
  const __bf16* arow1 = arow0 + (size_t)16 * K;
  const __bf16* bbase = Wt + (size_t)halfsel * OUT + n0 + (lane & 15);

  Frags f0, f1;
  ldfrags(f0, arow0, arow1, bbase, 0, OUT);

  int kb = 0;
  for (; kb + 64 < K; kb += 64) {
    ldfrags(f1, arow0, arow1, bbase, kb + 32, OUT);
    // Pull the weight stream into cache ~4 K-steps ahead (global_prefetch_b8).
    __builtin_prefetch(bbase + (size_t)(kb + 128) * OUT, 0, 1);
    do_wmma(f0, acc);
    ldfrags(f0, arow0, arow1, bbase, kb + 64, OUT);
    do_wmma(f1, acc);
  }
  // Tail: kb and kb+32 remain (kb + 64 == K).
  ldfrags(f1, arow0, arow1, bbase, kb + 32, OUT);
  do_wmma(f0, acc);
  do_wmma(f1, acc);
}

// GEMM1: h = A1 * Wt1; epilogue = gelu, then directly emit layer-2 augmented
// activations into A2[n, o*9 + j] (fuses away the 64 MB h buffer).
// blockIdx.x = OUT chunk (fast-varying) so A1 rows stay cache-resident.
__global__ __launch_bounds__(128)
void gemm_aug(const __bf16* __restrict__ A, const __bf16* __restrict__ Wt,
              __bf16* __restrict__ A2, int K, int OUT, int K2) {
  int lane = threadIdx.x & 31;
  int wave = threadIdx.x >> 5;
  int m0 = blockIdx.y * 128 + wave * 32;
  int n0 = blockIdx.x * 64;
  v8f acc[2][4] = {};
  wmma_mainloop(A, Wt, K, OUT, m0, n0, lane, acc);

  // C layout: acc[g][t][r] -> row m0 + g*16 + r + 8*(lane>>4),
  //                           col n0 + t*16 + (lane&15)
  int rowoff = (lane >> 4) * 8;
  int colb = n0 + (lane & 15);
#pragma unroll
  for (int g = 0; g < 2; ++g) {
#pragma unroll
    for (int t = 0; t < 4; ++t) {
      int o = colb + t * 16;
#pragma unroll
      for (int r = 0; r < 8; ++r) {
        int m = m0 + g * 16 + rowoff + r;
        float hv = geluf(acc[g][t][r]);
        __bf16* dst = A2 + (size_t)m * K2 + (size_t)o * NAUG;
        dst[0] = (__bf16)siluf(hv);
        float b[11];
        bspline8(hv, b);
#pragma unroll
        for (int j = 0; j < NCOEF; ++j) dst[1 + j] = (__bf16)b[j];
      }
    }
  }
}

// GEMM2: out = A2 * Wt2 + x (residual), f32 output.
__global__ __launch_bounds__(128)
void gemm_resid(const __bf16* __restrict__ A, const __bf16* __restrict__ Wt,
                const float* __restrict__ X, float* __restrict__ Out,
                int K, int OUT) {
  int lane = threadIdx.x & 31;
  int wave = threadIdx.x >> 5;
  int m0 = blockIdx.y * 128 + wave * 32;
  int n0 = blockIdx.x * 64;
  v8f acc[2][4] = {};
  wmma_mainloop(A, Wt, K, OUT, m0, n0, lane, acc);

  int rowoff = (lane >> 4) * 8;
  int colb = n0 + (lane & 15);
#pragma unroll
  for (int g = 0; g < 2; ++g) {
#pragma unroll
    for (int t = 0; t < 4; ++t) {
      int o = colb + t * 16;
#pragma unroll
      for (int r = 0; r < 8; ++r) {
        size_t idx = (size_t)(m0 + g * 16 + rowoff + r) * OUT + o;
        Out[idx] = acc[g][t][r] + X[idx];
      }
    }
  }
}

extern "C" void kernel_launch(void* const* d_in, const int* in_sizes, int n_in,
                              void* d_out, int out_size, void* d_ws,
                              size_t ws_size, hipStream_t stream) {
  const float* x         = (const float*)d_in[0];
  const float* base_w1   = (const float*)d_in[1];
  const float* spline_w1 = (const float*)d_in[2];
  const float* scaler1   = (const float*)d_in[3];
  const float* base_w2   = (const float*)d_in[4];
  const float* spline_w2 = (const float*)d_in[5];
  const float* scaler2   = (const float*)d_in[6];
  // d_in[7], d_in[8] = grids: uniform & known at compile time, not needed.
  float* out = (float*)d_out;

  const int N = 4096, D = 1024, DFF = 4096;
  const int K1 = D * NAUG;    // 9216
  const int K2 = DFF * NAUG;  // 36864

  char* ws = (char*)d_ws;
  __bf16* Wt1 = (__bf16*)ws; ws += (size_t)K1 * DFF * sizeof(__bf16);
  __bf16* Wt2 = (__bf16*)ws; ws += (size_t)K2 * D * sizeof(__bf16);
  __bf16* A1  = (__bf16*)ws; ws += (size_t)N * K1 * sizeof(__bf16);
  __bf16* A2  = (__bf16*)ws;

  // Weight packing (per call; deterministic, no caching allowed).
  pack_wt<<<dim3(DFF / 256, D), 256, 0, stream>>>(base_w1, spline_w1, scaler1,
                                                  Wt1, D, DFF);
  pack_wt<<<dim3(D / 256, DFF), 256, 0, stream>>>(base_w2, spline_w2, scaler2,
                                                  Wt2, DFF, D);
  // Layer-1 augmented activations from x.
  augment_act<<<dim3(D / 256, N), 256, 0, stream>>>(x, A1, D);
  // Layer 1 GEMM + GELU + layer-2 augmentation (fused epilogue).
  // blockIdx.x = DFF chunks (fast) so the A1 row-slice stays L2-resident.
  gemm_aug<<<dim3(DFF / 64, N / 128), 128, 0, stream>>>(A1, Wt1, A2, K1, DFF,
                                                        K2);
  // Layer 2 GEMM + residual. blockIdx.x = D chunks (fast): the 302 MB A2
  // operand streams through HBM exactly once; Wt2 (75 MB) stays L2-resident.
  gemm_resid<<<dim3(D / 64, N / 128), 128, 0, stream>>>(A2, Wt2, x, out, K2, D);
}
